// NanoGPTBlock_41497974014415
// MI455X (gfx1250) — matrix operand order repaired and checked by reading, not compiled, not run on hardware
//
#include <hip/hip_runtime.h>
#include <hip/hip_bf16.h>
#include <math.h>
#include <stdint.h>

#define D_MODEL 768
#define SEQ     4096
#define NBATCH  4
#define MTOT    (NBATCH * SEQ)

typedef _Float16 half_t;
typedef __attribute__((ext_vector_type(16))) _Float16 v16h;
typedef __attribute__((ext_vector_type(8)))  _Float16 v8h;
typedef __attribute__((ext_vector_type(8)))  float    v8f;

union FragH {
    v16h v;
    v8h  h[2];
};

static __device__ __forceinline__ v8f wmma16x16x32(v16h a, v16h b, v8f c) {
    // D = A(16x32 f16) * B(32x16 f16) + C(16x16 f32)
    return __builtin_amdgcn_wmma_f32_16x16x32_f16(
        /*neg_a=*/false, a, /*neg_b=*/false, b,
        /*c_mod=*/(short)0, c, /*reuse_a=*/false, /*reuse_b=*/false);
}

// Async copy of one 32-byte segment (global -> LDS) using the CDNA5
// GLOBAL_LOAD_ASYNC_TO_LDS path (tracked by ASYNCcnt).  INST_OFFSET is added
// to both LDS and global addresses, so offset:16 covers the second b128.
static __device__ __forceinline__ void async_copy32(void* lds_ptr, const void* gptr) {
    unsigned ldsa = (unsigned)(uintptr_t)lds_ptr;       // LDS aperture: offset in addr[31:0]
    unsigned long long ga = (unsigned long long)(uintptr_t)gptr;
    asm volatile("global_load_async_to_lds_b128 %0, %1, off\n\t"
                 "global_load_async_to_lds_b128 %0, %1, off offset:16"
                 :: "v"(ldsa), "v"(ga) : "memory");
}

static __device__ __forceinline__ void wait_async_all() {
    asm volatile("s_wait_asynccnt 0x0" ::: "memory");
}

// ---------------------------------------------------------------------------
// Elementwise fp32 -> f16 convert (for fc1_w / fc2_w which are already [N][K])
// ---------------------------------------------------------------------------
__global__ __launch_bounds__(256) void cvt_kernel(const float* __restrict__ in,
                                                  half_t* __restrict__ out, int n) {
    int i = blockIdx.x * 256 + threadIdx.x;
    if (i < n) out[i] = (half_t)in[i];
}

// ---------------------------------------------------------------------------
// fp32 [K][N] -> f16 [N][K] transpose-convert (for wq/wk/wv)
// ---------------------------------------------------------------------------
__global__ __launch_bounds__(256) void cvt_transpose_kernel(const float* __restrict__ W,
                                                            half_t* __restrict__ Wt) {
    __shared__ float tile[32][33];
    const int k0 = blockIdx.x * 32, d0 = blockIdx.y * 32;
    const int c = threadIdx.x;
#pragma unroll
    for (int i = 0; i < 4; ++i) {
        const int r = threadIdx.y + i * 8;
        tile[r][c] = W[(size_t)(k0 + r) * D_MODEL + d0 + c];
    }
    __syncthreads();
#pragma unroll
    for (int i = 0; i < 4; ++i) {
        const int r = threadIdx.y + i * 8;
        Wt[(size_t)(d0 + r) * D_MODEL + k0 + c] = (half_t)tile[c][r];
    }
}

// ---------------------------------------------------------------------------
// f16 V [B*T][D] -> f16 Vt [B][D][T]   (so A@V B-fragments are contiguous)
// ---------------------------------------------------------------------------
__global__ __launch_bounds__(256) void transpose_v_kernel(const half_t* __restrict__ V,
                                                          half_t* __restrict__ Vt) {
    __shared__ half_t tile[32][33];
    const int b = blockIdx.z;
    const int t0 = blockIdx.x * 32, d0 = blockIdx.y * 32;
    const int c = threadIdx.x;
#pragma unroll
    for (int i = 0; i < 4; ++i) {
        const int r = threadIdx.y + i * 8;
        tile[r][c] = V[((size_t)b * SEQ + t0 + r) * D_MODEL + d0 + c];
    }
    __syncthreads();
#pragma unroll
    for (int i = 0; i < 4; ++i) {
        const int r = threadIdx.y + i * 8;
        Vt[((size_t)b * D_MODEL + d0 + r) * SEQ + t0 + c] = tile[c][r];
    }
}

// ---------------------------------------------------------------------------
// LayerNorm faithful to reference bug: out = (x - mu/sqrt(var)) * g + b,
// var with ddof=1.  One 256-thread block per row of 768.  Output f16.
// ---------------------------------------------------------------------------
__global__ __launch_bounds__(256) void ln_kernel(const float* __restrict__ x,
                                                 const float* __restrict__ g,
                                                 const float* __restrict__ b,
                                                 half_t* __restrict__ out) {
    __shared__ float s1[256];
    __shared__ float s2[256];
    const int row = blockIdx.x;
    const float* xr = x + (size_t)row * D_MODEL;
    const int t = threadIdx.x;
    float v0 = xr[t], v1 = xr[t + 256], v2 = xr[t + 512];
    s1[t] = v0 + v1 + v2;
    s2[t] = v0 * v0 + v1 * v1 + v2 * v2;
    __syncthreads();
    for (int o = 128; o > 0; o >>= 1) {
        if (t < o) { s1[t] += s1[t + o]; s2[t] += s2[t + o]; }
        __syncthreads();
    }
    const float sum = s1[0], sq = s2[0];
    const float mu  = sum * (1.0f / D_MODEL);
    const float var = (sq - sum * mu) * (1.0f / (D_MODEL - 1));
    const float sub = mu * rsqrtf(var);   // mu / sqrt(var)  (buggy-LN faithful)
    half_t* orow = out + (size_t)row * D_MODEL;
    orow[t]       = (half_t)((v0 - sub) * g[t]       + b[t]);
    orow[t + 256] = (half_t)((v1 - sub) * g[t + 256] + b[t + 256]);
    orow[t + 512] = (half_t)((v2 - sub) * g[t + 512] + b[t + 512]);
}

// ---------------------------------------------------------------------------
// WMMA GEMM: C[M][N] = A[M][768] * Bt[N][768]^T  (+bias, +relu, +residual)
// Block tile 128x256, 8 waves (2x4), wave tile 64x64 = 4x4 WMMA tiles.
// Staging: double-buffered LDS filled with GLOBAL_LOAD_ASYNC_TO_LDS_B128
// (ASYNCcnt), prefetching tile k+1 while computing tile k.
// ---------------------------------------------------------------------------
__global__ __launch_bounds__(256) void gemm_kernel(const half_t* __restrict__ A,
                                                   const half_t* __restrict__ Bt,
                                                   const float* __restrict__ bias,
                                                   const float* __restrict__ resid,
                                                   float* __restrict__ Cf,
                                                   half_t* __restrict__ Ch,
                                                   int relu) {
    __shared__ __align__(16) half_t Al[2][128][40];  // +8 half pad: conflict-free b128
    __shared__ __align__(16) half_t Bl[2][256][40];
    const int tid = threadIdx.x, lane = tid & 31, w = tid >> 5;
    const int wm = w >> 2, wn = w & 3;               // 2x4 wave grid
    const int m_blk = blockIdx.x * 128, n_blk = blockIdx.y * 256;
    const int fr = lane & 15, hi = lane >> 4;

    const v8f vzero = {};
    v8f acc[4][4];
#pragma unroll
    for (int i = 0; i < 4; ++i)
#pragma unroll
        for (int j = 0; j < 4; ++j) acc[i][j] = vzero;

    const int arow = tid >> 1, aseg = (tid & 1) * 16;

    auto stage = [&](int buf, int k0) {
        // A tile: 128 rows x 32 halves (256 segments of 32B, one per thread)
        async_copy32(&Al[buf][arow][aseg],
                     A + (size_t)(m_blk + arow) * D_MODEL + k0 + aseg);
        // Bt tile: 256 rows x 32 halves (512 segments, two per thread)
#pragma unroll
        for (int i = 0; i < 2; ++i) {
            const int idx = tid + i * 256;
            const int brow = idx >> 1, bseg = (idx & 1) * 16;
            async_copy32(&Bl[buf][brow][bseg],
                         Bt + (size_t)(n_blk + brow) * D_MODEL + k0 + bseg);
        }
    };

    stage(0, 0);
    wait_async_all();
    __syncthreads();

    for (int k0 = 0; k0 < D_MODEL; k0 += 32) {
        const int cur = (k0 >> 5) & 1;
        if (k0 + 32 < D_MODEL) stage(cur ^ 1, k0 + 32);  // prefetch next tile

        FragH af[4];
        const int akb = hi * 8;   // A: halves 0..7 -> K=akb.., halves 8..15 -> K=akb+16..
#pragma unroll
        for (int mt = 0; mt < 4; ++mt) {
            const int r = wm * 64 + mt * 16 + fr;
            af[mt].h[0] = *(const v8h*)&Al[cur][r][akb];
            af[mt].h[1] = *(const v8h*)&Al[cur][r][akb + 16];
        }
        const int bkb = hi * 8;   // B: halves 0..7 -> K=bkb.., halves 8..15 -> K=bkb+16..
#pragma unroll
        for (int nt = 0; nt < 4; ++nt) {
            FragH bf;
            const int r = wn * 64 + nt * 16 + fr;
            bf.h[0] = *(const v8h*)&Bl[cur][r][bkb];
            bf.h[1] = *(const v8h*)&Bl[cur][r][bkb + 16];
#pragma unroll
            for (int mt = 0; mt < 4; ++mt)
                acc[mt][nt] = wmma16x16x32(af[mt].v, bf.v, acc[mt][nt]);
        }
        wait_async_all();   // prefetched tile fully in LDS (this wave's share)
        __syncthreads();    // ...and everyone else's; cur buffer free to reuse
    }

#pragma unroll
    for (int mt = 0; mt < 4; ++mt)
#pragma unroll
        for (int nt = 0; nt < 4; ++nt)
#pragma unroll
            for (int g = 0; g < 8; ++g) {
                const int gr = m_blk + wm * 64 + mt * 16 + g + 8 * hi;
                const int gc = n_blk + wn * 64 + nt * 16 + fr;
                float v = acc[mt][nt][g];
                if (bias)  v += bias[gc];
                if (relu)  v = fmaxf(v, 0.0f);
                if (resid) v += resid[(size_t)gr * D_MODEL + gc];
                if (Cf) Cf[(size_t)gr * D_MODEL + gc] = v;
                if (Ch) Ch[(size_t)gr * D_MODEL + gc] = (half_t)v;
            }
}

// ---------------------------------------------------------------------------
// Flash-style attention. One WG = 32 query rows of one batch. 8 waves.
// Key tiles of 64.  Phase 1: each wave one 16x16 S tile (2x4 layout),
// 24 WMMA k-steps over D=768.  Online softmax (f32, LDS).  Phase 2: each
// wave owns O[32][96] (12 accum tiles), A@V via Ps(LDS) x Vt(global/L2).
// Q/K/Vt are 6 MB per batch: resident in the 192 MB L2, so fragments are
// loaded straight from global.
// ---------------------------------------------------------------------------
__global__ __launch_bounds__(256) void attn_kernel(const half_t* __restrict__ Q,
                                                   const half_t* __restrict__ K,
                                                   const half_t* __restrict__ Vt,
                                                   float* __restrict__ O) {
    __shared__ float            Sf[32][68];
    __shared__ __align__(16) half_t Ps[32][72];
    __shared__ float mrow[32], lrow[32], crow[32];

    const int tid = threadIdx.x, lane = tid & 31, w = tid >> 5;
    const int fr = lane & 15, hi = lane >> 4;
    const int b = blockIdx.x >> 7;            // 128 query-blocks per batch
    const int q0 = (blockIdx.x & 127) * 32;
    const int smt = w >> 2, snt = w & 3;      // S tile coords: 2 x 4
    const float scale = 0.03608439182435161f; // 1/sqrt(768)

    if (tid < 32) { mrow[tid] = -3.0e38f; lrow[tid] = 0.0f; }
    const v8f vzero = {};
    v8f oacc[2][6];
#pragma unroll
    for (int i = 0; i < 2; ++i)
#pragma unroll
        for (int j = 0; j < 6; ++j) oacc[i][j] = vzero;

    const size_t qrow = ((size_t)b * SEQ + q0 + smt * 16 + fr) * D_MODEL;
    __syncthreads();

    for (int kt = 0; kt < SEQ; kt += 64) {
        // ---- Phase 1: S = (Q Kt) * scale for this wave's 16x16 tile ----
        v8f s = vzero;
        const size_t krow = ((size_t)b * SEQ + kt + snt * 16 + fr) * D_MODEL;
        for (int k0 = 0; k0 < D_MODEL; k0 += 32) {
            FragH a, bb;
            a.h[0]  = *(const v8h*)(Q + qrow + k0 + hi * 8);
            a.h[1]  = *(const v8h*)(Q + qrow + k0 + hi * 8 + 16);
            bb.h[0] = *(const v8h*)(K + krow + k0 + hi * 8);
            bb.h[1] = *(const v8h*)(K + krow + k0 + hi * 8 + 16);
            s = wmma16x16x32(a.v, bb.v, s);
        }
#pragma unroll
        for (int g = 0; g < 8; ++g)
            Sf[smt * 16 + g + 8 * hi][snt * 16 + fr] = s[g] * scale;
        __syncthreads();

        // ---- Online softmax: one thread per query row ----
        if (tid < 32) {
            const int r = tid;
            float mo = mrow[r], mx = mo;
            for (int c = 0; c < 64; ++c) mx = fmaxf(mx, Sf[r][c]);
            const float cr = __expf(mo - mx);
            float ls = lrow[r] * cr;
            for (int c = 0; c < 64; ++c) {
                const float p = __expf(Sf[r][c] - mx);
                Ps[r][c] = (half_t)p;
                ls += p;
            }
            mrow[r] = mx; lrow[r] = ls; crow[r] = cr;
        }
        __syncthreads();

        // ---- Phase 2: rescale O, then O += P @ Vtile ----
        float cv[2][8];
#pragma unroll
        for (int mt = 0; mt < 2; ++mt)
#pragma unroll
            for (int g = 0; g < 8; ++g) cv[mt][g] = crow[mt * 16 + g + 8 * hi];
#pragma unroll
        for (int mt = 0; mt < 2; ++mt)
#pragma unroll
            for (int nt = 0; nt < 6; ++nt)
#pragma unroll
                for (int g = 0; g < 8; ++g) oacc[mt][nt][g] *= cv[mt][g];

#pragma unroll
        for (int ks = 0; ks < 64; ks += 32) {
            FragH pa[2], vb[6];
#pragma unroll
            for (int mt = 0; mt < 2; ++mt) {
                pa[mt].h[0] = *(const v8h*)&Ps[mt * 16 + fr][ks + hi * 8];
                pa[mt].h[1] = *(const v8h*)&Ps[mt * 16 + fr][ks + hi * 8 + 16];
            }
#pragma unroll
            for (int nt = 0; nt < 6; ++nt) {
                const int col = w * 96 + nt * 16 + fr;
                const half_t* vp = Vt + ((size_t)b * D_MODEL + col) * SEQ + kt + ks + hi * 8;
                vb[nt].h[0] = *(const v8h*)vp;
                vb[nt].h[1] = *(const v8h*)(vp + 16);
            }
#pragma unroll
            for (int mt = 0; mt < 2; ++mt)
#pragma unroll
                for (int nt = 0; nt < 6; ++nt)
                    oacc[mt][nt] = wmma16x16x32(pa[mt].v, vb[nt].v, oacc[mt][nt]);
        }
        __syncthreads();
    }

    // ---- Finalize: O /= l, store f32 ----
    float rv[2][8];
#pragma unroll
    for (int mt = 0; mt < 2; ++mt)
#pragma unroll
        for (int g = 0; g < 8; ++g) rv[mt][g] = 1.0f / lrow[mt * 16 + g + 8 * hi];
#pragma unroll
    for (int mt = 0; mt < 2; ++mt)
#pragma unroll
        for (int nt = 0; nt < 6; ++nt)
#pragma unroll
            for (int g = 0; g < 8; ++g) {
                const size_t gr = (size_t)b * SEQ + q0 + mt * 16 + g + 8 * hi;
                const int gc = w * 96 + nt * 16 + fr;
                O[gr * D_MODEL + gc] = oacc[mt][nt][g] * rv[mt][g];
            }
}

// ---------------------------------------------------------------------------
extern "C" void kernel_launch(void* const* d_in, const int* in_sizes, int n_in,
                              void* d_out, int out_size, void* d_ws, size_t ws_size,
                              hipStream_t stream) {
    (void)in_sizes; (void)n_in; (void)out_size; (void)ws_size;
    const float* x     = (const float*)d_in[0];
    const float* ln1_g = (const float*)d_in[1];
    const float* ln1_b = (const float*)d_in[2];
    const float* wq    = (const float*)d_in[3];
    const float* wk    = (const float*)d_in[4];
    const float* wv    = (const float*)d_in[5];
    const float* ln2_g = (const float*)d_in[6];
    const float* ln2_b = (const float*)d_in[7];
    const float* fc1_w = (const float*)d_in[8];
    const float* fc1_b = (const float*)d_in[9];
    const float* fc2_w = (const float*)d_in[10];
    const float* fc2_b = (const float*)d_in[11];
    float* out = (float*)d_out;

    char* ws = (char*)d_ws;
    const size_t S16 = (size_t)MTOT * D_MODEL * sizeof(half_t);  // 24 MB
    const size_t S32 = (size_t)MTOT * D_MODEL * sizeof(float);   // 48 MB
    const size_t SW  = (size_t)D_MODEL * D_MODEL;
    half_t* yh  = (half_t*)(ws);              // slot0: yh, later zh
    half_t* qh  = (half_t*)(ws + S16);        // slot1: qh, later hh
    half_t* kh  = (half_t*)(ws + 2 * S16);
    half_t* vh  = (half_t*)(ws + 3 * S16);
    half_t* vt  = (half_t*)(ws + 4 * S16);
    float*  at  = (float*) (ws + 5 * S16);    // attention out (f32)
    half_t* wqT = (half_t*)(ws + 5 * S16 + S32);
    half_t* wkT = wqT + SW;
    half_t* wvT = wkT + SW;
    half_t* f1h = wvT + SW;
    half_t* f2h = f1h + SW;
    half_t* zh = yh;   // reuse after QKV done
    half_t* hh = qh;   // reuse after attention done

    const dim3 t256(256);
    const dim3 bT(32, 8);
    const dim3 gT(D_MODEL / 32, D_MODEL / 32);
    const dim3 gG(MTOT / 128, D_MODEL / 256);
    const dim3 gV(SEQ / 32, D_MODEL / 32, NBATCH);
    const int nW = D_MODEL * D_MODEL;

    // Weight prep
    cvt_transpose_kernel<<<gT, bT, 0, stream>>>(wq, wqT);
    cvt_transpose_kernel<<<gT, bT, 0, stream>>>(wk, wkT);
    cvt_transpose_kernel<<<gT, bT, 0, stream>>>(wv, wvT);
    cvt_kernel<<<(nW + 255) / 256, t256, 0, stream>>>(fc1_w, f1h, nW);
    cvt_kernel<<<(nW + 255) / 256, t256, 0, stream>>>(fc2_w, f2h, nW);

    // LN1
    ln_kernel<<<MTOT, t256, 0, stream>>>(x, ln1_g, ln1_b, yh);

    // QKV projections (f16 out)
    gemm_kernel<<<gG, t256, 0, stream>>>(yh, wqT, nullptr, nullptr, nullptr, qh, 0);
    gemm_kernel<<<gG, t256, 0, stream>>>(yh, wkT, nullptr, nullptr, nullptr, kh, 0);
    gemm_kernel<<<gG, t256, 0, stream>>>(yh, wvT, nullptr, nullptr, nullptr, vh, 0);

    // V -> Vt [B][D][T]
    transpose_v_kernel<<<gV, bT, 0, stream>>>(vh, vt);

    // Attention (f32 out)
    attn_kernel<<<NBATCH * SEQ / 32, t256, 0, stream>>>(qh, kh, vt, at);

    // LN2
    ln_kernel<<<MTOT, t256, 0, stream>>>(at, ln2_g, ln2_b, zh);

    // MLP: fc1 (+bias, relu) -> hh f16 ; fc2 (+bias, +residual x) -> d_out f32
    gemm_kernel<<<gG, t256, 0, stream>>>(zh, f1h, fc1_b, nullptr, nullptr, hh, 1);
    gemm_kernel<<<gG, t256, 0, stream>>>(hh, f2h, fc2_b, x, out, nullptr, 0);
}
